// SCBottleneckBlock_8607114461666
// MI455X (gfx1250) — matrix operand order repaired and checked by reading, not compiled, not run on hardware
//
#include <hip/hip_runtime.h>

// ---------------------------------------------------------------------------
// SCBottleneck sparse-conv block on gfx1250 via bf16 WMMA (16x16x32, f32 acc).
// N=100000 voxels, CIN=256, CMID=64, 27 neighbor offsets, BN folded in f32
// epilogues. All GEMMs run through v_wmma_f32_16x16x32_bf16.
// ---------------------------------------------------------------------------

#define NVOX  100000
#define CIN   256
#define CMID  64
#define NOFF  27
#define EPSB  1e-5f

typedef __attribute__((ext_vector_type(16))) __bf16 v16bf;
typedef __attribute__((ext_vector_type(8)))  float  v8f;

__device__ __forceinline__ unsigned f2bf(float f) {
    union { float f; unsigned u; } x; x.f = f;
    unsigned u = x.u;
    u += 0x7FFFu + ((u >> 16) & 1u);       // round-to-nearest-even
    return u >> 16;
}
__device__ __forceinline__ unsigned pack2bf(float lo, float hi) {
    return f2bf(lo) | (f2bf(hi) << 16);
}

// A fragment: 16x32 bf16 tile, a_words = packed K-pair words, row-major,
// row stride = rs words. Lane l<16: M=l, K in {0..7,16..23}; lanes 16..31:
// M=l-16, K in {8..15,24..31}.  word index = j + (j>=4?4:0) + 4*half.
__device__ __forceinline__ v16bf load_a_frag(const unsigned* a_words, int rs,
                                             int kword_base, int lane) {
    const int m = lane & 15, half = lane >> 4;
    union { unsigned u[8]; v16bf v; } r;
#pragma unroll
    for (int j = 0; j < 8; ++j) {
        int w = j + ((j >= 4) ? 4 : 0) + 4 * half;
        r.u[j] = a_words[m * rs + kword_base + w];
    }
    return r.v;
}

// B fragment: 32x16 bf16 tile from packed [Kpairs][ncols] words.
// lane l: N = l&15; half = l>>4 selects K {0..7,16..23} / {8..15,24..31}; VGPR j -> Kpair row.
__device__ __forceinline__ v16bf load_b_frag(const unsigned* b_words, int ncols,
                                             int kword_base, int col_base, int lane) {
    const int n = lane & 15, half = lane >> 4;
    union { unsigned u[8]; v16bf v; } r;
#pragma unroll
    for (int j = 0; j < 8; ++j) {
        int kr = kword_base + j + ((j >= 4) ? 4 : 0) + 4 * half;
        r.u[j] = b_words[kr * ncols + col_base + n];
    }
    return r.v;
}

__device__ __forceinline__ v8f wmma_bf16(v16bf a, v16bf b, v8f c) {
    return __builtin_amdgcn_wmma_f32_16x16x32_bf16(false, a, false, b,
                                                   (short)0, c, false, false);
}

// ---------------------------------------------------------------------------
// Prep: fp32 weights -> bf16 K-pair-packed words; zero sentinel row of h1p.
//   w1p: [128][64] words, w2p: 27 x [32][64] words, w3p: [32][256] words.
// ---------------------------------------------------------------------------
__global__ __launch_bounds__(256) void k_prep(
    const float* __restrict__ w1, const float* __restrict__ w2,
    const float* __restrict__ w3, unsigned* __restrict__ w1p,
    unsigned* __restrict__ w2p, unsigned* __restrict__ w3p,
    unsigned short* __restrict__ h1p) {
    const int i = blockIdx.x * blockDim.x + threadIdx.x;
    if (i < 128 * 64) {                       // w1: [256][64]
        int kp = i >> 6, n = i & 63;
        w1p[i] = pack2bf(w1[(2 * kp) * 64 + n], w1[(2 * kp + 1) * 64 + n]);
    }
    if (i < NOFF * 32 * 64) {                 // w2: [27][64][64]
        int k = i >> 11, j = i & 2047;
        int kp = j >> 6, n = j & 63;
        const float* wk = w2 + k * 64 * 64;
        w2p[i] = pack2bf(wk[(2 * kp) * 64 + n], wk[(2 * kp + 1) * 64 + n]);
    }
    if (i < 32 * 256) {                       // w3: [64][256]
        int kp = i >> 8, n = i & 255;
        w3p[i] = pack2bf(w3[(2 * kp) * 256 + n], w3[(2 * kp + 1) * 256 + n]);
    }
    if (i < CMID) h1p[(size_t)NVOX * CMID + i] = 0;   // sentinel zero row
}

// ---------------------------------------------------------------------------
// Layer1: h1 = relu(bn1(features @ w1)) -> bf16 [N+1][64]
// block = 256 thr (8 waves), M-tile 32, full K=256 in LDS.
// ---------------------------------------------------------------------------
__global__ __launch_bounds__(256) void k_layer1(
    const float* __restrict__ feat, const unsigned* __restrict__ w1p,
    const float* __restrict__ gg, const float* __restrict__ bb,
    const float* __restrict__ mm, const float* __restrict__ vv,
    unsigned short* __restrict__ h1p) {
    __shared__ unsigned sB[128 * 64];   // 32 KB packed w1
    __shared__ unsigned sA[32 * 128];   // 16 KB packed A tile (32 rows x 256 K)
    const int t = threadIdx.x, lane = t & 31, wave = t >> 5;
    const int row0 = blockIdx.x * 32;

    for (int i = t; i < 128 * 64; i += 256) sB[i] = w1p[i];
    const float2* f2 = (const float2*)feat;
#pragma unroll
    for (int q = 0; q < 16; ++q) {
        int w = t + 256 * q;                    // 4096 words total
        int r = w >> 7, kw = w & 127;
        float2 x = f2[(size_t)(row0 + r) * 128 + kw];
        sA[w] = pack2bf(x.x, x.y);
    }
    __syncthreads();

    const int wm = (wave >> 2) * 16, wn = (wave & 3) * 16;
    v8f c = {0.f, 0.f, 0.f, 0.f, 0.f, 0.f, 0.f, 0.f};
#pragma unroll
    for (int kk = 0; kk < 8; ++kk) {
        v16bf a = load_a_frag(&sA[wm * 128], 128, kk * 16, lane);
        v16bf b = load_b_frag(sB, 64, kk * 16, wn, lane);
        c = wmma_bf16(a, b, c);
    }
    const int n = lane & 15, half = lane >> 4;
    const int ch = wn + n;
    float sc = gg[ch] * rsqrtf(vv[ch] + EPSB);
    float bi = bb[ch] - mm[ch] * sc;
#pragma unroll
    for (int j = 0; j < 8; ++j) {
        int mr = wm + 8 * half + j;
        float val = fmaxf(0.f, c[j] * sc + bi);
        h1p[(size_t)(row0 + mr) * CMID + ch] = (unsigned short)f2bf(val);
    }
}

// ---------------------------------------------------------------------------
// Layer2: 27-offset gather-GEMM-accumulate, bn2+relu -> bf16 h2 [N][64]
// ---------------------------------------------------------------------------
__global__ __launch_bounds__(256) void k_layer2(
    const unsigned short* __restrict__ h1p, const int* __restrict__ nbr,
    const unsigned* __restrict__ w2p,
    const float* __restrict__ gg, const float* __restrict__ bb,
    const float* __restrict__ mm, const float* __restrict__ vv,
    unsigned short* __restrict__ h2) {
    __shared__ unsigned sB[32 * 64];    // 8 KB packed w2[k]
    __shared__ unsigned sA[32 * 32];    // 4 KB gathered rows (32 x 64 bf16)
    __shared__ int sIdx[32];
    const unsigned* h1w = (const unsigned*)h1p;   // 32 words per row
    const int t = threadIdx.x, lane = t & 31, wave = t >> 5;
    const int row0 = blockIdx.x * 32;
    const int wm = (wave >> 2) * 16, wn = (wave & 3) * 16;

    v8f c = {0.f, 0.f, 0.f, 0.f, 0.f, 0.f, 0.f, 0.f};
    for (int k = 0; k < NOFF; ++k) {
        __syncthreads();                         // LDS reuse fence
        if (t < 32) sIdx[t] = nbr[(size_t)k * NVOX + row0 + t];
#pragma unroll
        for (int q = 0; q < 8; ++q) sB[t + 256 * q] = w2p[k * 2048 + t + 256 * q];
        __syncthreads();
#pragma unroll
        for (int q = 0; q < 4; ++q) {            // gather 32 rows x 32 words
            int w = t + 256 * q;
            int r = w >> 5, kw = w & 31;
            sA[w] = h1w[(size_t)sIdx[r] * 32 + kw];
        }
        __syncthreads();
#pragma unroll
        for (int ks = 0; ks < 2; ++ks) {
            v16bf a = load_a_frag(&sA[wm * 32], 32, ks * 16, lane);
            v16bf b = load_b_frag(sB, 64, ks * 16, wn, lane);
            c = wmma_bf16(a, b, c);
        }
    }
    const int n = lane & 15, half = lane >> 4;
    const int ch = wn + n;
    float sc = gg[ch] * rsqrtf(vv[ch] + EPSB);
    float bi = bb[ch] - mm[ch] * sc;
#pragma unroll
    for (int j = 0; j < 8; ++j) {
        int mr = wm + 8 * half + j;
        float val = fmaxf(0.f, c[j] * sc + bi);
        h2[(size_t)(row0 + mr) * CMID + ch] = (unsigned short)f2bf(val);
    }
}

// ---------------------------------------------------------------------------
// Layer3: out = relu(relu(bn3(h2 @ w3)) + features)  -> fp32 [N][256]
// Each wave covers 4 N-tiles of its 16-col slot (64 of 256 cols per wave set).
// ---------------------------------------------------------------------------
__global__ __launch_bounds__(256) void k_layer3(
    const unsigned short* __restrict__ h2, const unsigned* __restrict__ w3p,
    const float* __restrict__ feat,
    const float* __restrict__ gg, const float* __restrict__ bb,
    const float* __restrict__ mm, const float* __restrict__ vv,
    float* __restrict__ out) {
    __shared__ unsigned sB[32 * 256];   // 32 KB packed w3
    __shared__ unsigned sA[32 * 32];    // 4 KB A tile
    const unsigned* h2w = (const unsigned*)h2;
    const int t = threadIdx.x, lane = t & 31, wave = t >> 5;
    const int row0 = blockIdx.x * 32;

    for (int i = t; i < 32 * 256; i += 256) sB[i] = w3p[i];
#pragma unroll
    for (int q = 0; q < 4; ++q) {
        int w = t + 256 * q;
        int r = w >> 5, kw = w & 31;
        sA[w] = h2w[(size_t)(row0 + r) * 32 + kw];
    }
    __syncthreads();

    const int wm = (wave >> 2) * 16, wn = (wave & 3) * 16;
    const v8f z8 = {0.f, 0.f, 0.f, 0.f, 0.f, 0.f, 0.f, 0.f};
    v8f acc[4] = {z8, z8, z8, z8};
#pragma unroll
    for (int tt = 0; tt < 4; ++tt) {
        int cbase = wn + tt * 64;
#pragma unroll
        for (int ks = 0; ks < 2; ++ks) {
            v16bf a = load_a_frag(&sA[wm * 32], 32, ks * 16, lane);
            v16bf b = load_b_frag(sB, 256, ks * 16, cbase, lane);
            acc[tt] = wmma_bf16(a, b, acc[tt]);
        }
    }
    const int n = lane & 15, half = lane >> 4;
#pragma unroll
    for (int tt = 0; tt < 4; ++tt) {
        int ch = wn + tt * 64 + n;
        float sc = gg[ch] * rsqrtf(vv[ch] + EPSB);
        float bi = bb[ch] - mm[ch] * sc;
#pragma unroll
        for (int j = 0; j < 8; ++j) {
            int mr = wm + 8 * half + j;
            size_t o = (size_t)(row0 + mr) * CIN + ch;
            float val = fmaxf(0.f, acc[tt][j] * sc + bi);
            out[o] = fmaxf(0.f, val + feat[o]);
        }
    }
}

// ---------------------------------------------------------------------------
extern "C" void kernel_launch(void* const* d_in, const int* in_sizes, int n_in,
                              void* d_out, int out_size, void* d_ws, size_t ws_size,
                              hipStream_t stream) {
    const float* feat = (const float*)d_in[0];
    const int*   nbr  = (const int*)d_in[1];
    const float* w1   = (const float*)d_in[2];
    const float* g1   = (const float*)d_in[3];
    const float* b1   = (const float*)d_in[4];
    const float* m1   = (const float*)d_in[5];
    const float* v1   = (const float*)d_in[6];
    const float* w2   = (const float*)d_in[7];
    const float* g2   = (const float*)d_in[8];
    const float* b2   = (const float*)d_in[9];
    const float* m2   = (const float*)d_in[10];
    const float* v2   = (const float*)d_in[11];
    const float* w3   = (const float*)d_in[12];
    const float* g3   = (const float*)d_in[13];
    const float* b3   = (const float*)d_in[14];
    const float* m3   = (const float*)d_in[15];
    const float* v3   = (const float*)d_in[16];
    float* out = (float*)d_out;

    char* ws = (char*)d_ws;
    unsigned* w1p = (unsigned*)(ws + 0);                       //  32768 B
    unsigned* w2p = (unsigned*)(ws + 32768);                   // 221184 B
    unsigned* w3p = (unsigned*)(ws + 32768 + 221184);          //  32768 B
    unsigned short* h1p = (unsigned short*)(ws + 286720);      // (N+1)*64*2
    unsigned short* h2  = (unsigned short*)(ws + 286720 + 12800256);

    const int prep_threads = NOFF * 32 * 64;                   // 55296 (max range)
    k_prep<<<(prep_threads + 255) / 256, 256, 0, stream>>>(w1, w2, w3, w1p, w2p,
                                                           w3p, h1p);
    const int nblk = NVOX / 32;                                // 3125, exact
    k_layer1<<<nblk, 256, 0, stream>>>(feat, w1p, g1, b1, m1, v1, h1p);
    k_layer2<<<nblk, 256, 0, stream>>>(h1p, nbr, w2p, g2, b2, m2, v2, h2);
    k_layer3<<<nblk, 256, 0, stream>>>(h2, w3p, feat, g3, b3, m3, v3, out);
}
